// RNN_22995254903422
// MI455X (gfx1250) — matrix operand order repaired and checked by reading, not compiled, not run on hardware
//
#include <hip/hip_runtime.h>

// Problem constants (from the reference): T=4096, IN=2048, H=2048, OUT=1
#define T_SEQ 4096
#define DIN   2048
#define DH    2048
#define G4H   8192      // 4*H
#define NUM_WG 64       // persistent workgroups in the recurrent kernel

typedef __attribute__((ext_vector_type(16))) __bf16 v16bf;
typedef __attribute__((ext_vector_type(8)))  __bf16 v8bf;
typedef __attribute__((ext_vector_type(4)))  __bf16 v4bf;
typedef __attribute__((ext_vector_type(8)))  float  v8f;

__device__ __forceinline__ v16bf combine16(v8bf lo, v8bf hi) {
  v16bf r;
#pragma unroll
  for (int i = 0; i < 8; ++i) { r[i] = lo[i]; r[i + 8] = hi[i]; }
  return r;
}

// Gate-row permutation: permuted row p = 16*w + 4*q + r holds original row
// q*H + 4*w + r  (q = gate i/f/g/o, w = cell group, r = cell-in-group).
// This makes one WMMA wave own all 4 gates of its own 4 LSTM cells.
__device__ __forceinline__ size_t orig_row(int p) {
  const int q = (p >> 2) & 3;
  const int w = p >> 4;
  const int r = p & 3;
  return (size_t)q * DH + (w << 2) + r;
}

// ---------------------------------------------------------------------------
// fp32 -> bf16 conversion (grid-strided)
// ---------------------------------------------------------------------------
__global__ void cvt_f32_bf16(const float* __restrict__ in,
                             __bf16* __restrict__ out, long n) {
  long i = (long)blockIdx.x * blockDim.x + threadIdx.x;
  long stride = (long)gridDim.x * blockDim.x;
  for (; i < n; i += stride) out[i] = (__bf16)in[i];
}

// fp32 -> bf16 with gate-row permutation (K = 2048 elements per row)
__global__ void cvt_rows_perm(const float* __restrict__ in,
                              __bf16* __restrict__ out, long n) {
  long i = (long)blockIdx.x * blockDim.x + threadIdx.x;
  long stride = (long)gridDim.x * blockDim.x;
  for (; i < n; i += stride) {
    const int  p = (int)(i >> 11);       // row (K = 2048 = 1<<11)
    const int  k = (int)(i & 2047);
    out[i] = (__bf16)in[orig_row(p) * DIN + k];
  }
}

// ---------------------------------------------------------------------------
// init: h_bf16 = 0, y[t] = b_out, barrier counters = 0
// ---------------------------------------------------------------------------
__global__ void init_state(__bf16* __restrict__ h_b, float* __restrict__ y,
                           const float* __restrict__ b_out,
                           unsigned* __restrict__ bar) {
  int i = blockIdx.x * blockDim.x + threadIdx.x;
  if (i < DH)    h_b[i] = (__bf16)0.0f;
  if (i < T_SEQ) y[i] = b_out[0];
  if (i == 0) { bar[0] = 0u; bar[1] = 0u; }
}

// ---------------------------------------------------------------------------
// Gx[t, p] = sum_k X[t,k] * W_ih[orig_row(p), k] + biases   (permuted cols)
// One wave computes a 16(M) x 64(N) tile, K-loop software-pipelined.
// ---------------------------------------------------------------------------
__global__ void __launch_bounds__(256)
lstm_gatex_wmma(const __bf16* __restrict__ Xb,      // [T, IN]
                const __bf16* __restrict__ Wihp,    // [4H, IN] (rows permuted)
                const float*  __restrict__ b_ih,    // [4H] (original order)
                const float*  __restrict__ b_hh,    // [4H]
                float* __restrict__ Gx) {           // [T, 4H] (permuted cols)
  const int wid  = (blockIdx.x * blockDim.x + threadIdx.x) >> 5;
  const int lane = threadIdx.x & 31;
  const int half = lane >> 4;          // 0: lanes 0-15, 1: lanes 16-31
  const int l16  = lane & 15;

  const int tileM = (wid >> 7) << 4;   // (wid / 128) * 16
  const int n0    = (wid & 127) << 6;  // (wid % 128) * 64

  const __bf16* aRow = Xb   + (size_t)(tileM + l16) * DIN + half * 16;
  const __bf16* bRow = Wihp + (size_t)(n0 + l16) * DIN + half * 16;

  v8f acc0 = {}, acc1 = {}, acc2 = {}, acc3 = {};

  // ---- software-pipelined K loop (double-buffered fragments) ----
  v8bf  alo = *(const v8bf*)(aRow);
  v8bf  ahi = *(const v8bf*)(aRow + 8);
  v16bf b0  = *(const v16bf*)(bRow);
  v16bf b1  = *(const v16bf*)(bRow + (size_t)16 * DIN);
  v16bf b2  = *(const v16bf*)(bRow + (size_t)32 * DIN);
  v16bf b3  = *(const v16bf*)(bRow + (size_t)48 * DIN);

  for (int k0 = 32; k0 < DIN; k0 += 32) {
    __builtin_prefetch((const void*)(aRow + k0 + 32), 0, 3);
    __builtin_prefetch((const void*)(bRow + k0 + 32), 0, 3);
    v8bf  alo_n = *(const v8bf*)(aRow + k0);
    v8bf  ahi_n = *(const v8bf*)(aRow + k0 + 8);
    v16bf b0_n  = *(const v16bf*)(bRow + k0);
    v16bf b1_n  = *(const v16bf*)(bRow + (size_t)16 * DIN + k0);
    v16bf b2_n  = *(const v16bf*)(bRow + (size_t)32 * DIN + k0);
    v16bf b3_n  = *(const v16bf*)(bRow + (size_t)48 * DIN + k0);

    v16bf a = combine16(alo, ahi);
    acc0 = __builtin_amdgcn_wmma_f32_16x16x32_bf16(false, a, false, b0, (short)0, acc0, false, false);
    acc1 = __builtin_amdgcn_wmma_f32_16x16x32_bf16(false, a, false, b1, (short)0, acc1, false, false);
    acc2 = __builtin_amdgcn_wmma_f32_16x16x32_bf16(false, a, false, b2, (short)0, acc2, false, false);
    acc3 = __builtin_amdgcn_wmma_f32_16x16x32_bf16(false, a, false, b3, (short)0, acc3, false, false);

    alo = alo_n; ahi = ahi_n; b0 = b0_n; b1 = b1_n; b2 = b2_n; b3 = b3_n;
  }
  {
    v16bf a = combine16(alo, ahi);
    acc0 = __builtin_amdgcn_wmma_f32_16x16x32_bf16(false, a, false, b0, (short)0, acc0, false, false);
    acc1 = __builtin_amdgcn_wmma_f32_16x16x32_bf16(false, a, false, b1, (short)0, acc1, false, false);
    acc2 = __builtin_amdgcn_wmma_f32_16x16x32_bf16(false, a, false, b2, (short)0, acc2, false, false);
    acc3 = __builtin_amdgcn_wmma_f32_16x16x32_bf16(false, a, false, b3, (short)0, acc3, false, false);
  }

  // D layout: VGPR r -> (M = r + 8*half, N = l16). Bias looked up through the
  // inverse permutation (Gx itself stays in permuted column order).
  const int mBase = tileM + half * 8;
#pragma unroll
  for (int s = 0; s < 4; ++s) {
    const int colp = n0 + s * 16 + l16;          // permuted column
    const size_t c = orig_row(colp);             // original gate row
    const float bias = b_ih[c] + b_hh[c];
    v8f acc = (s == 0) ? acc0 : (s == 1) ? acc1 : (s == 2) ? acc2 : acc3;
    float* gout = Gx + (size_t)mBase * G4H + colp;
#pragma unroll
    for (int r = 0; r < 8; ++r)
      gout[(size_t)r * G4H] = acc[r] + bias;
  }
}

// ---------------------------------------------------------------------------
// grid-wide barrier for the persistent recurrent kernel.
// The cluster-barrier is an architected NOP when not dispatched as a cluster
// (ClusterID==0) and a fast HW barrier when it is.
// ---------------------------------------------------------------------------
__device__ __forceinline__ void grid_barrier(unsigned* cnt, unsigned* gen) {
  __syncthreads();
  __builtin_amdgcn_s_cluster_barrier();
  if (threadIdx.x == 0) {
    __threadfence();
    unsigned g = __hip_atomic_load(gen, __ATOMIC_RELAXED, __HIP_MEMORY_SCOPE_AGENT);
    if (atomicAdd(cnt, 1u) == NUM_WG - 1) {
      __hip_atomic_store(cnt, 0u, __ATOMIC_RELAXED, __HIP_MEMORY_SCOPE_AGENT);
      __threadfence();
      __hip_atomic_fetch_add(gen, 1u, __ATOMIC_RELEASE, __HIP_MEMORY_SCOPE_AGENT);
    } else {
      while (__hip_atomic_load(gen, __ATOMIC_ACQUIRE, __HIP_MEMORY_SCOPE_AGENT) == g)
        __builtin_amdgcn_s_sleep(1);
    }
  }
  __syncthreads();
}

// ---------------------------------------------------------------------------
// Persistent recurrent kernel: 64 WGs x 256 threads = 512 waves.
// Wave w owns permuted gate rows [16w, 16w+16) = gates {i,f,g,o} of cells
// j = 4w..4w+3. Every lane loads the same h chunk into B, so all 16 B columns
// equal h and every lane's D fragment holds the GEMV results (replicated):
//   lanes 0-15 : VGPR r -> row 16w+r   (r=0..3: i, r=4..7: f)
//   lanes 16-31: VGPR r -> row 16w+8+r (r=0..3: g, r=4..7: o)
// One __shfl set + local cell update + 8B h store + ONE grid barrier/step.
// ---------------------------------------------------------------------------
__global__ void __launch_bounds__(256, 1)
lstm_recurrent(const __bf16* __restrict__ Whhp,   // [4H, H] (rows permuted)
               const float*  __restrict__ Gx,     // [T, 4H] (permuted cols)
               const float*  __restrict__ Wout,   // [H]
               __bf16* __restrict__ h_b,          // [H]
               float* __restrict__ y,             // [T], pre-set to b_out
               unsigned* __restrict__ bar) {      // [2]: count, generation
  const int tid  = blockIdx.x * 256 + threadIdx.x;
  const int lane = threadIdx.x & 31;
  const int wid  = tid >> 5;           // 0..511
  const int row0 = wid << 4;
  const int half = lane >> 4;
  const int l16  = lane & 15;

  const __bf16* wrow = Whhp + (size_t)(row0 + l16) * DH + half * 16;
  const __bf16* hb   = h_b + half * 16;           // K 0..15 / 16..31 per half
  const float*  wo   = Wout + (wid << 2);

  __shared__ float ypart[8];

  float c0 = 0.0f, c1 = 0.0f, c2 = 0.0f, c3 = 0.0f;

  for (int t = 0; t < T_SEQ; ++t) {
    // ---- gate GEMV via WMMA (software-pipelined K loop) ----
    v8f acc = {};
    v8bf  alo = *(const v8bf*)(wrow);
    v8bf  ahi = *(const v8bf*)(wrow + 8);
    v16bf b   = *(const v16bf*)(hb);
    for (int k0 = 32; k0 < DH; k0 += 32) {
      __builtin_prefetch((const void*)(wrow + k0 + 32), 0, 3);
      v8bf  alo_n = *(const v8bf*)(wrow + k0);
      v8bf  ahi_n = *(const v8bf*)(wrow + k0 + 8);
      v16bf b_n   = *(const v16bf*)(hb + k0);
      acc = __builtin_amdgcn_wmma_f32_16x16x32_bf16(false, combine16(alo, ahi),
                                                    false, b, (short)0, acc, false, false);
      alo = alo_n; ahi = ahi_n; b = b_n;
    }
    acc = __builtin_amdgcn_wmma_f32_16x16x32_bf16(false, combine16(alo, ahi),
                                                  false, b, (short)0, acc, false, false);

    // add precomputed input gates (broadcast loads; per-half rows)
    const float* gx = Gx + (size_t)t * G4H + row0 + half * 8;
#pragma unroll
    for (int r = 0; r < 8; ++r) acc[r] += gx[r];

    // bring half-1 results (g, o) to lane 0
    float sh[8];
#pragma unroll
    for (int r = 0; r < 8; ++r) sh[r] = __shfl(acc[r], 16, 32);

    if (lane == 0) {
      // acc[0..3]=i, acc[4..7]=f, sh[0..3]=g, sh[4..7]=o for cells 4*wid+r
      float ysum = 0.0f;
      v4bf hv;
#pragma unroll
      for (int r = 0; r < 4; ++r) {
        float si = 1.0f / (1.0f + __expf(-acc[r]));
        float sf = 1.0f / (1.0f + __expf(-acc[4 + r]));
        float tg = tanhf(sh[r]);
        float so = 1.0f / (1.0f + __expf(-sh[4 + r]));
        float& c = (r == 0) ? c0 : (r == 1) ? c1 : (r == 2) ? c2 : c3;
        c = sf * c + si * tg;
        float h = so * tanhf(c);
        hv[r] = (__bf16)h;
        ysum += h * wo[r];
      }
      *(v4bf*)(h_b + (wid << 2)) = hv;   // 8-byte store of 4 bf16
      ypart[threadIdx.x >> 5] = ysum;
    }
    __syncthreads();
    if (threadIdx.x == 0) {
      float s = 0.0f;
#pragma unroll
      for (int i = 0; i < 8; ++i) s += ypart[i];
      atomicAdd(&y[t], s);
    }
    grid_barrier(&bar[0], &bar[1]);     // h_b now visible grid-wide
  }
}

// ---------------------------------------------------------------------------
// Host-side launcher
// ---------------------------------------------------------------------------
extern "C" void kernel_launch(void* const* d_in, const int* in_sizes, int n_in,
                              void* d_out, int out_size, void* d_ws, size_t ws_size,
                              hipStream_t stream) {
  const float* x     = (const float*)d_in[0];  // [T,1,IN]
  const float* W_ih  = (const float*)d_in[1];  // [4H, IN]
  const float* W_hh  = (const float*)d_in[2];  // [4H, H]
  const float* b_ih  = (const float*)d_in[3];  // [4H]
  const float* b_hh  = (const float*)d_in[4];  // [4H]
  const float* W_out = (const float*)d_in[5];  // [1, H]
  const float* b_out = (const float*)d_in[6];  // [1]
  float* y = (float*)d_out;                    // [T]

  char* ws = (char*)d_ws;
  size_t off = 0;
  auto alloc = [&](size_t bytes) -> void* {
    void* p = ws + off;
    off = (off + bytes + 255) & ~(size_t)255;
    return p;
  };

  __bf16*   Xb   = (__bf16*)alloc((size_t)T_SEQ * DIN * 2);  // 16.8 MB
  __bf16*   Wihp = (__bf16*)alloc((size_t)G4H * DIN * 2);    // 33.6 MB
  __bf16*   Whhp = (__bf16*)alloc((size_t)G4H * DH * 2);     // 33.6 MB
  float*    Gx   = (float*)alloc((size_t)T_SEQ * G4H * 4);   // 134 MB
  __bf16*   h_b  = (__bf16*)alloc((size_t)DH * 2);
  unsigned* bar  = (unsigned*)alloc(256);

  // 1) precision conversion fp32 -> bf16; weights get the gate-row permutation
  cvt_f32_bf16 <<<1024, 256, 0, stream>>>(x,    Xb,   (long)T_SEQ * DIN);
  cvt_rows_perm<<<1024, 256, 0, stream>>>(W_ih, Wihp, (long)G4H * DIN);
  cvt_rows_perm<<<1024, 256, 0, stream>>>(W_hh, Whhp, (long)G4H * DH);

  // 2) state init (h=0, y[t]=b_out, barrier reset)
  init_state<<<32, 256, 0, stream>>>(h_b, y, b_out, bar);

  // 3) big parallel WMMA GEMM: Gx = X @ W_ih^T + b_ih + b_hh (permuted cols)
  lstm_gatex_wmma<<<4096, 256, 0, stream>>>(Xb, Wihp, b_ih, b_hh, Gx);

  // 4) persistent sequential LSTM loop: WMMA GEMV + fused cell update,
  //    one grid barrier per timestep
  lstm_recurrent<<<NUM_WG, 256, 0, stream>>>(Whhp, Gx, W_out, h_b, y, bar);
}
